// KaNCD_RGCN_fine1_91044716740748
// MI455X (gfx1250) — matrix operand order repaired and checked by reading, not compiled, not run on hardware
//
#include <hip/hip_runtime.h>
#include <hip/hip_bf16.h>

#define DEVINL __device__ __forceinline__

typedef _Float16 half_t;
typedef __attribute__((ext_vector_type(8)))  _Float16 v8h;
typedef __attribute__((ext_vector_type(16))) _Float16 v16h;
typedef __attribute__((ext_vector_type(8)))  float    v8f;

static constexpr int S_   = 50000;
static constexpr int E_   = 20000;
static constexpr int Kn_  = 64;      // knowledge concepts
static constexpr int D_   = 64;      // embedding dim
static constexpr int B_   = 32768;   // batch
static constexpr int NNZ_ = 1000000;

DEVINL float sigmoidf_(float x) { return 1.0f / (1.0f + __expf(-x)); }

// ---- WMMA operand loaders (wave32, contiguous-K order, same for A and B) -------------------
// A 16x32 f16, row-major source [*, lda]: lane l -> row (l&15),
// 16 consecutive K starting at 16*(l>>4) (contiguous 32B).
DEVINL v16h load_a_tile(const half_t* __restrict__ A, int lda, int row0, int k0, int lane) {
  const int hs = lane >> 4;
  const int r  = lane & 15;
  const half_t* p = A + (size_t)(row0 + r) * lda + k0 + 16 * hs;
  return *(const v16h*)(p);
}

// B 32x16 f16 from N-major storage BT[N][Kd]: lane l -> col (l&15),
// 16 consecutive K starting at k0 + 16*(l>>4) (contiguous 32B).
DEVINL v16h load_b_tile(const half_t* __restrict__ BT, int ldb, int n0, int k0, int lane) {
  const int hs = lane >> 4;
  const int n  = n0 + (lane & 15);
  return *(const v16h*)(BT + (size_t)n * ldb + k0 + 16 * hs);
}

// ---- prep kernels --------------------------------------------------------
__global__ void f32_to_f16(const float* __restrict__ src, half_t* __restrict__ dst, int n) {
  int i = blockIdx.x * blockDim.x + threadIdx.x;
  if (i < n) dst[i] = (half_t)src[i];
}

// dst[n*Kd + k] = relu(src[k*Nd + n])  (PosLinear weight, pre-transposed to N-major)
__global__ void transpose_relu_f16(const float* __restrict__ src, half_t* __restrict__ dst,
                                   int Kd, int Nd) {
  int i = blockIdx.x * blockDim.x + threadIdx.x;
  if (i >= Kd * Nd) return;
  int n = i / Kd, k = i % Kd;
  dst[i] = (half_t)fmaxf(src[(size_t)k * Nd + n], 0.0f);
}

// ---- embedding GEMM: sigmoid(A[rows,64] @ know^T) -> out0 and outsum -----
__global__ __launch_bounds__(256)
void embed_gemm_sig(const half_t* __restrict__ A, const half_t* __restrict__ knowT,
                    float* __restrict__ out0, float* __restrict__ outsum, int nStrips) {
  const int lane  = threadIdx.x & 31;
  const int strip = blockIdx.x * (blockDim.x >> 5) + (threadIdx.x >> 5);
  if (strip >= nStrips) return;                       // wave-uniform exit
  const int row0 = strip * 16;
  const v16h a0 = load_a_tile(A, D_, row0, 0,  lane);
  const v16h a1 = load_a_tile(A, D_, row0, 32, lane);
  const int hs = lane >> 4, nl = lane & 15;
#pragma unroll
  for (int n0 = 0; n0 < Kn_; n0 += 16) {
    v8f c = {};
    v16h b0 = load_b_tile(knowT, D_, n0, 0,  lane);
    v16h b1 = load_b_tile(knowT, D_, n0, 32, lane);
    c = __builtin_amdgcn_wmma_f32_16x16x32_f16(false, a0, false, b0, (short)0, c, false, false);
    c = __builtin_amdgcn_wmma_f32_16x16x32_f16(false, a1, false, b1, (short)0, c, false, false);
#pragma unroll
    for (int i = 0; i < 8; ++i) {
      const float v = sigmoidf_(c[i]);
      const size_t o = (size_t)(row0 + i + 8 * hs) * Kn_ + n0 + nl;
      out0[o]   = v;
      outsum[o] = v;
    }
  }
}

// ---- SpMM scatter-add: out[row] += val * X[col]  (16 lanes/edge, float4) -
__global__ __launch_bounds__(256)
void spmm_atomic(const int* __restrict__ idx, const float* __restrict__ val,
                 const float* __restrict__ X, float* __restrict__ out, int nnz) {
  int g = blockIdx.x * blockDim.x + threadIdx.x;
  int e = g >> 4;
  if (e >= nnz) return;
  // speculative prefetch of the edge stream (global_prefetch_b8)
  __builtin_prefetch(idx + e + 4096, 0, 0);
  const int j   = (g & 15) << 2;
  const int row = idx[e];
  const int col = idx[nnz + e];
  const float v = val[e];
  const float4 xv = *(const float4*)(X + (size_t)col * Kn_ + j);
  float* o = out + (size_t)row * Kn_ + j;
  atomicAdd(o + 0, v * xv.x);
  atomicAdd(o + 1, v * xv.y);
  atomicAdd(o + 2, v * xv.z);
  atomicAdd(o + 3, v * xv.w);
}

// ---- x = disc[e] * ((statsum[s] - diffsum[e]) / 3) * kn_emb --------------
__global__ void build_x(const float* __restrict__ statsum, const float* __restrict__ diffsum,
                        const float* __restrict__ disc_W, const float* __restrict__ kn_emb,
                        const int* __restrict__ stu_id, const int* __restrict__ exer_id,
                        half_t* __restrict__ x, int total) {
  int i = blockIdx.x * blockDim.x + threadIdx.x;
  if (i >= total) return;
  const int b = i >> 6, j = i & 63;
  const int s = stu_id[b], e = exer_id[b];
  const float d = sigmoidf_(disc_W[e]);
  const float v = d * ((statsum[(size_t)s * Kn_ + j] - diffsum[(size_t)e * Kn_ + j]) * (1.0f / 3.0f))
                    * kn_emb[i];
  x[i] = (half_t)v;
}

// ---- MLP layer: out = sigmoid(A[M,KD] @ BT^T + bias), f16 in/out ---------
template<int KD, int NOUT>
__global__ __launch_bounds__(256)
void mlp_gemm_sig(const half_t* __restrict__ A, const half_t* __restrict__ BT,
                  const float* __restrict__ bias, half_t* __restrict__ out, int nStrips) {
  const int lane  = threadIdx.x & 31;
  const int strip = blockIdx.x * (blockDim.x >> 5) + (threadIdx.x >> 5);
  if (strip >= nStrips) return;                       // wave-uniform exit
  const int row0 = strip * 16;
  constexpr int KT = KD / 32;
  v16h aReg[KT];
#pragma unroll
  for (int kt = 0; kt < KT; ++kt) aReg[kt] = load_a_tile(A, KD, row0, kt * 32, lane);
  const int hs = lane >> 4, nl = lane & 15;
#pragma unroll
  for (int n0 = 0; n0 < NOUT; n0 += 16) {
    v8f c = {};
#pragma unroll
    for (int kt = 0; kt < KT; ++kt) {
      v16h b = load_b_tile(BT, KD, n0, kt * 32, lane);
      c = __builtin_amdgcn_wmma_f32_16x16x32_f16(false, aReg[kt], false, b, (short)0, c, false, false);
    }
    const float bv = bias[n0 + nl];
#pragma unroll
    for (int i = 0; i < 8; ++i) {
      out[(size_t)(row0 + i + 8 * hs) * NOUT + n0 + nl] = (half_t)sigmoidf_(c[i] + bv);
    }
  }
}

// ---- final N=1 layer (VALU dot; WMMA wasteful at N=1) --------------------
__global__ void final_layer(const half_t* __restrict__ h2, const float* __restrict__ W3,
                            const float* __restrict__ b3, float* __restrict__ out, int n) {
  int b = blockIdx.x * blockDim.x + threadIdx.x;
  if (b >= n) return;
  float acc = b3[0];
  const half_t* row = h2 + (size_t)b * 128;
#pragma unroll 8
  for (int j = 0; j < 128; ++j) acc += (float)row[j] * fmaxf(W3[j], 0.0f);
  out[b] = sigmoidf_(acc);
}

// --------------------------------------------------------------------------
extern "C" void kernel_launch(void* const* d_in, const int* in_sizes, int n_in,
                              void* d_out, int out_size, void* d_ws, size_t ws_size,
                              hipStream_t stream) {
  (void)in_sizes; (void)n_in; (void)out_size; (void)ws_size;

  const float* stu_W  = (const float*)d_in[0];
  const float* exer_W = (const float*)d_in[1];
  const float* know_W = (const float*)d_in[2];
  const float* disc_W = (const float*)d_in[3];
  const float* W1 = (const float*)d_in[4];
  const float* b1 = (const float*)d_in[5];
  const float* W2 = (const float*)d_in[6];
  const float* b2 = (const float*)d_in[7];
  const float* W3 = (const float*)d_in[8];
  const float* b3 = (const float*)d_in[9];
  const int*   ui1_idx = (const int*)d_in[10];  const float* ui1_val = (const float*)d_in[11];
  const int*   ui0_idx = (const int*)d_in[12];  const float* ui0_val = (const float*)d_in[13];
  const int*   iu1_idx = (const int*)d_in[14];  const float* iu1_val = (const float*)d_in[15];
  const int*   iu0_idx = (const int*)d_in[16];  const float* iu0_val = (const float*)d_in[17];
  const float* kn_emb  = (const float*)d_in[18];
  const int*   stu_id  = (const int*)d_in[19];
  const int*   exer_id = (const int*)d_in[20];
  float* out = (float*)d_out;

  // workspace carve-out (256B-aligned slabs), ~75 MB total
  char* ws = (char*)d_ws;
  size_t off = 0;
  auto alloc = [&](size_t bytes) -> char* {
    char* p = ws + off;
    off = (off + bytes + 255) & ~(size_t)255;
    return p;
  };
  half_t* stu_h   = (half_t*)alloc((size_t)S_ * D_ * 2);
  half_t* exer_h  = (half_t*)alloc((size_t)E_ * D_ * 2);
  half_t* know_h  = (half_t*)alloc((size_t)Kn_ * D_ * 2);
  half_t* w1t_h   = (half_t*)alloc((size_t)256 * 64 * 2);     // relu(W1)^T, N-major
  half_t* w2t_h   = (half_t*)alloc((size_t)128 * 256 * 2);    // relu(W2)^T, N-major
  float*  stat0   = (float*)alloc((size_t)S_ * Kn_ * 4);
  float*  diff0   = (float*)alloc((size_t)E_ * Kn_ * 4);
  float*  statsum = (float*)alloc((size_t)S_ * Kn_ * 4);
  float*  diffsum = (float*)alloc((size_t)E_ * Kn_ * 4);
  half_t* x_h     = (half_t*)alloc((size_t)B_ * 64 * 2);
  half_t* h1_h    = (half_t*)alloc((size_t)B_ * 256 * 2);
  half_t* h2_h    = (half_t*)alloc((size_t)B_ * 128 * 2);

  const int T = 256;
  auto grid1 = [&](long long n) { return (unsigned)((n + T - 1) / T); };

  // 1) precision conversion + weight transpose/relu
  f32_to_f16<<<grid1((long long)S_ * D_), T, 0, stream>>>(stu_W,  stu_h,  S_ * D_);
  f32_to_f16<<<grid1((long long)E_ * D_), T, 0, stream>>>(exer_W, exer_h, E_ * D_);
  f32_to_f16<<<grid1(Kn_ * D_),           T, 0, stream>>>(know_W, know_h, Kn_ * D_);
  transpose_relu_f16<<<grid1(64 * 256),  T, 0, stream>>>(W1, w1t_h, 64, 256);
  transpose_relu_f16<<<grid1(256 * 128), T, 0, stream>>>(W2, w2t_h, 256, 128);

  // 2) stat0 / diff0 (sigmoid GEMMs) — also seed the accumulators
  {
    const int strips_s = S_ / 16, strips_e = E_ / 16;      // 3125, 1250
    embed_gemm_sig<<<(strips_s + 7) / 8, T, 0, stream>>>(stu_h,  know_h, stat0, statsum, strips_s);
    embed_gemm_sig<<<(strips_e + 7) / 8, T, 0, stream>>>(exer_h, know_h, diff0, diffsum, strips_e);
  }

  // 3) four SpMM scatter-adds (L2-resident on MI455X's 192MB L2)
  {
    const unsigned g = grid1((long long)NNZ_ * 16);
    spmm_atomic<<<g, T, 0, stream>>>(ui1_idx, ui1_val, diff0, statsum, NNZ_);
    spmm_atomic<<<g, T, 0, stream>>>(ui0_idx, ui0_val, diff0, statsum, NNZ_);
    spmm_atomic<<<g, T, 0, stream>>>(iu1_idx, iu1_val, stat0, diffsum, NNZ_);
    spmm_atomic<<<g, T, 0, stream>>>(iu0_idx, iu0_val, stat0, diffsum, NNZ_);
  }

  // 4) x = disc * ((statsum - diffsum)/3) * kn_emb  (folds the /3 normalization)
  build_x<<<grid1((long long)B_ * 64), T, 0, stream>>>(statsum, diffsum, disc_W, kn_emb,
                                                       stu_id, exer_id, x_h, B_ * 64);

  // 5) MLP with WMMA (f32 accumulate, fused sigmoid+bias, f16 activations)
  {
    const int strips_b = B_ / 16;                           // 2048
    mlp_gemm_sig<64, 256><<<(strips_b + 7) / 8, T, 0, stream>>>(x_h,  w1t_h, b1, h1_h, strips_b);
    mlp_gemm_sig<256, 128><<<(strips_b + 7) / 8, T, 0, stream>>>(h1_h, w2t_h, b2, h2_h, strips_b);
  }

  // 6) final N=1 layer
  final_layer<<<grid1(B_), T, 0, stream>>>(h2_h, W3, b3, out, B_);
}
